// CausalCrossAttention_13640816132183
// MI455X (gfx1250) — matrix operand order repaired and checked by reading, not compiled, not run on hardware
//
#include <hip/hip_runtime.h>

// ---------------------------------------------------------------------------
// Types & helpers
// ---------------------------------------------------------------------------
typedef __attribute__((ext_vector_type(16))) __bf16 v16bf;
typedef __attribute__((ext_vector_type(8)))  float  v8f;

struct Bytes32 { uint4 lo, hi; };

static __device__ __forceinline__ v16bf make_frag(uint4 lo, uint4 hi) {
  Bytes32 t{lo, hi};
  return __builtin_bit_cast(v16bf, t);
}

static __device__ __forceinline__ unsigned short f2bf(float f) {
  unsigned int u = __float_as_uint(f);
  unsigned int r = (u + 0x7FFFu + ((u >> 16) & 1u)) >> 16; // RNE
  return (unsigned short)r;
}

static __device__ __forceinline__ void store_out(unsigned short* p, int i, float v) {
  p[i] = f2bf(v);
}
static __device__ __forceinline__ void store_out(float* p, int i, float v) {
  __builtin_nontemporal_store(v, p + i);   // final output: never reread
}

// ---------------------------------------------------------------------------
// fp32 -> bf16 conversion
// ---------------------------------------------------------------------------
__global__ void f32_to_bf16_k(const float* __restrict__ in,
                              unsigned short* __restrict__ out, int n) {
  int i = blockIdx.x * blockDim.x + threadIdx.x;
  if (i < n) out[i] = f2bf(in[i]);
}

// ---------------------------------------------------------------------------
// C[M,N] = A[M,K](bf16) @ W[N,K](bf16)^T + bias  (torch Linear layout)
// Block: 128 threads = 4 waves stacked in M. Wave tile 64x64: 4x4 WMMA
// accumulators, K stepped 64 per iteration (2 sub-steps folded to immediate
// offsets). All addressing is kernarg-base + running 32-bit element offsets
// so loads lower to GLOBAL (not FLAT) vector memory ops.
// ---------------------------------------------------------------------------
template <typename OutT>
__global__ __launch_bounds__(128) void gemm_bias_wmma(
    const unsigned short* __restrict__ A,  // [M,K]
    const unsigned short* __restrict__ W,  // [N,K]
    const float* __restrict__ bias,        // [N]
    OutT* __restrict__ C,                  // [M,N]
    int M, int N, int K) {
  const int lane = threadIdx.x & 31;
  const int wave = threadIdx.x >> 5;
  const int half = lane >> 4;   // which 16-lane half
  const int l16  = lane & 15;

  const int rowBase = blockIdx.y * 256 + wave * 64;
  const int colBase = blockIdx.x * 64;

  // Running per-lane element offsets (A: K = 8h..+7 / 16+8h..+7 chunks;
  // B: lane n+16h holds K = 8h..+7 / 16+8h..+7 of W row n)
  int offA[4], offB[4];
#pragma unroll
  for (int mt = 0; mt < 4; ++mt)
    offA[mt] = (rowBase + mt * 16 + l16) * K + 8 * half;
#pragma unroll
  for (int nt = 0; nt < 4; ++nt)
    offB[nt] = (colBase + nt * 16 + l16) * K + 8 * half;

  v8f acc[4][4] = {};

  for (int k0 = 0; k0 < K; k0 += 64) {
    __builtin_prefetch(A + offA[0] + 512, 0, 0);   // ~1KB ahead in this row
    __builtin_prefetch(W + offB[0] + 512, 0, 0);
#pragma unroll
    for (int ks = 0; ks < 64; ks += 32) {          // folds into load immediates
      v16bf af[4], bf[4];
#pragma unroll
      for (int mt = 0; mt < 4; ++mt)
        af[mt] = make_frag(*(const uint4*)(A + offA[mt] + ks),
                           *(const uint4*)(A + offA[mt] + ks + 16));
#pragma unroll
      for (int nt = 0; nt < 4; ++nt)
        bf[nt] = make_frag(*(const uint4*)(W + offB[nt] + ks),
                           *(const uint4*)(W + offB[nt] + ks + 16));
#pragma unroll
      for (int mt = 0; mt < 4; ++mt)
#pragma unroll
        for (int nt = 0; nt < 4; ++nt)
          acc[mt][nt] = __builtin_amdgcn_wmma_f32_16x16x32_bf16(
              false, af[mt], false, bf[nt], (short)0, acc[mt][nt], false, false);
    }
#pragma unroll
    for (int t = 0; t < 4; ++t) { offA[t] += 64; offB[t] += 64; }
  }

  // Epilogue: row offset walks down, bias hoisted per column tile.
  float bv[4];
#pragma unroll
  for (int nt = 0; nt < 4; ++nt) bv[nt] = bias[colBase + nt * 16 + l16];
#pragma unroll
  for (int mt = 0; mt < 4; ++mt) {
    int offC = (rowBase + mt * 16 + 8 * half) * N + colBase + l16;
#pragma unroll
    for (int j = 0; j < 8; ++j) {
#pragma unroll
      for (int nt = 0; nt < 4; ++nt)
        store_out(C, offC + nt * 16, acc[mt][nt][j] + bv[nt]);
      offC += N;
    }
  }
}

// ---------------------------------------------------------------------------
// Flash attention (causal), head dim 64, q-tile 64 (16 rows per wave),
// kv tiles of 64. Q/K/V/O are bf16 [B*S, D] with head slice at col h*64.
// ---------------------------------------------------------------------------
__global__ __launch_bounds__(128) void flash_attn_wmma(
    const unsigned short* __restrict__ Q,
    const unsigned short* __restrict__ Km,
    const unsigned short* __restrict__ Vm,
    unsigned short* __restrict__ O,
    int S, int H, int Dm) {
  constexpr int DH = 64;
  __shared__ __align__(16) unsigned short Vt[64 * 64];      // V^T tile [d][kv]
  __shared__ __align__(16) unsigned short Pb[4][16 * 64];   // per-wave P tile

  const int lane = threadIdx.x & 31;
  const int wave = threadIdx.x >> 5;
  const int half = lane >> 4;
  const int l16  = lane & 15;

  const int qt = blockIdx.x;            // q tile (64 rows)
  const int b  = blockIdx.y / H;
  const int h  = blockIdx.y % H;

  const int rowOff = b * S;             // row offset into [B*S]
  const int colOff = h * DH;
  const int kvStep = 64 * Dm;           // elements per kv tile

  // Preload Q A-fragments (2 K-chunks of 32 covering head dim 64)
  v16bf qf[2];
  {
    const int q0 = (rowOff + qt * 64 + wave * 16 + l16) * Dm + colOff;
#pragma unroll
    for (int kc = 0; kc < 2; ++kc) {
      const int p = q0 + kc * 32 + 8 * half;
      qf[kc] = make_frag(*(const uint4*)(Q + p), *(const uint4*)(Q + p + 16));
    }
  }

  // Running element offsets: K B-fragments (per column tile), V staging rows.
  int offK[4];
#pragma unroll
  for (int nt = 0; nt < 4; ++nt)
    offK[nt] = (rowOff + nt * 16 + l16) * Dm + colOff + 8 * half;
  const int vr = threadIdx.x >> 1;          // kv row 0..63
  const int vc = (threadIdx.x & 1) * 32;    // d column chunk
  int offV = (rowOff + vr) * Dm + colOff + vc;

  v8f oacc[4] = {};
  float mrow[8], lrow[8];
#pragma unroll
  for (int j = 0; j < 8; ++j) { mrow[j] = -3.0e38f; lrow[j] = 0.f; }

  const float scale = 0.125f;  // 1/sqrt(64)

  for (int kt = 0; kt <= qt; ++kt) {
    __syncthreads();  // all waves done with previous Vt
    // Stage V tile transposed into LDS: Vt[d][kv]; this thread owns
    // V[row vr][cols vc..vc+31].
    {
      uint4 dd[4];
      dd[0] = *(const uint4*)(Vm + offV);
      dd[1] = *(const uint4*)(Vm + offV + 8);
      dd[2] = *(const uint4*)(Vm + offV + 16);
      dd[3] = *(const uint4*)(Vm + offV + 24);
      const unsigned short* e = (const unsigned short*)dd;
#pragma unroll
      for (int t = 0; t < 32; ++t) Vt[(vc + t) * 64 + vr] = e[t];
    }
    __syncthreads();

    // Scores: S = Q @ K^T  (B-frag columns are K rows -> contiguous in global)
    v8f s[4] = {};
#pragma unroll
    for (int nt = 0; nt < 4; ++nt) {
#pragma unroll
      for (int kc = 0; kc < 2; ++kc) {
        const int p = offK[nt] + kc * 32;
        v16bf kf = make_frag(*(const uint4*)(Km + p), *(const uint4*)(Km + p + 16));
        s[nt] = __builtin_amdgcn_wmma_f32_16x16x32_bf16(
            false, qf[kc], false, kf, (short)0, s[nt], false, false);
      }
    }

    // Scale + causal mask + online softmax (rows live across 16-lane halves)
#pragma unroll
    for (int j = 0; j < 8; ++j) {
      const int qg = qt * 64 + wave * 16 + j + 8 * half;
      float tmax = -3.0e38f;
#pragma unroll
      for (int nt = 0; nt < 4; ++nt) {
        const int kg = kt * 64 + nt * 16 + l16;
        float v = s[nt][j] * scale;
        v = (kg > qg) ? -3.0e38f : v;
        s[nt][j] = v;
        tmax = fmaxf(tmax, v);
      }
#pragma unroll
      for (int msk = 1; msk < 16; msk <<= 1)
        tmax = fmaxf(tmax, __shfl_xor(tmax, msk, 32));
      const float mnew = fmaxf(mrow[j], tmax);
      const float alpha = __expf(mrow[j] - mnew);
      mrow[j] = mnew;
      float psum = 0.f;
#pragma unroll
      for (int nt = 0; nt < 4; ++nt) {
        const float pv2 = __expf(s[nt][j] - mnew);
        s[nt][j] = pv2;
        psum += pv2;
      }
#pragma unroll
      for (int msk = 1; msk < 16; msk <<= 1)
        psum += __shfl_xor(psum, msk, 32);
      lrow[j] = lrow[j] * alpha + psum;
#pragma unroll
      for (int dt = 0; dt < 4; ++dt) oacc[dt][j] *= alpha;
    }

    // Repack P (C-layout) -> A-layout via wave-private LDS
#pragma unroll
    for (int nt = 0; nt < 4; ++nt)
#pragma unroll
      for (int j = 0; j < 8; ++j)
        Pb[wave][(j + 8 * half) * 64 + nt * 16 + l16] = f2bf(s[nt][j]);

    // O += P @ V  (B-frags from transposed Vt, contiguous ds reads)
#pragma unroll
    for (int kc = 0; kc < 2; ++kc) {
      const unsigned short* pp = &Pb[wave][l16 * 64 + kc * 32 + 8 * half];
      v16bf pf = make_frag(*(const uint4*)pp, *(const uint4*)(pp + 16));
#pragma unroll
      for (int dt = 0; dt < 4; ++dt) {
        const unsigned short* vp = &Vt[(dt * 16 + l16) * 64 + kc * 32 + 8 * half];
        v16bf vf = make_frag(*(const uint4*)vp, *(const uint4*)(vp + 16));
        oacc[dt] = __builtin_amdgcn_wmma_f32_16x16x32_bf16(
            false, pf, false, vf, (short)0, oacc[dt], false, false);
      }
    }

    // advance running offsets to next kv tile
#pragma unroll
    for (int nt = 0; nt < 4; ++nt) offK[nt] += kvStep;
    offV += kvStep;
  }

  // Normalize + store (row offset walks down)
  int offO = (rowOff + qt * 64 + wave * 16 + 8 * half) * Dm + colOff + l16;
#pragma unroll
  for (int j = 0; j < 8; ++j) {
    const float rl = 1.0f / lrow[j];
#pragma unroll
    for (int dt = 0; dt < 4; ++dt) O[offO + dt * 16] = f2bf(oacc[dt][j] * rl);
    offO += Dm;
  }
}

// ---------------------------------------------------------------------------
// Host-side orchestration
// ---------------------------------------------------------------------------
extern "C" void kernel_launch(void* const* d_in, const int* in_sizes, int n_in,
                              void* d_out, int out_size, void* d_ws, size_t ws_size,
                              hipStream_t stream) {
  (void)in_sizes; (void)n_in; (void)out_size; (void)ws_size;

  constexpr int Bn = 2, S = 2048, D = 1024, H = 16;
  constexpr int M = Bn * S;                 // 4096 rows
  constexpr size_t NQ = (size_t)M * D;      // activation elements
  constexpr size_t NW = (size_t)D * D;      // weight elements

  const float* query = (const float*)d_in[0];
  const float* keyv  = (const float*)d_in[1];
  const float* Wq = (const float*)d_in[2];  const float* bq = (const float*)d_in[3];
  const float* Wk = (const float*)d_in[4];  const float* bk = (const float*)d_in[5];
  const float* Wv = (const float*)d_in[6];  const float* bv = (const float*)d_in[7];
  const float* Wo = (const float*)d_in[8];  const float* bo = (const float*)d_in[9];
  float* out = (float*)d_out;

  unsigned short* w   = (unsigned short*)d_ws;
  unsigned short* qb  = w;              // [M,D] bf16 query
  unsigned short* kvb = qb  + NQ;       // [M,D] bf16 key_value
  unsigned short* wqb = kvb + NQ;       // [D,D] bf16 weights
  unsigned short* wkb = wqb + NW;
  unsigned short* wvb = wkb + NW;
  unsigned short* wob = wvb + NW;
  unsigned short* Qp  = wob + NW;       // projected Q [M,D]
  unsigned short* Kp  = Qp  + NQ;
  unsigned short* Vp  = Kp  + NQ;
  unsigned short* Oat = Vp  + NQ;       // attention output [M,D]

  // 1) convert inputs to bf16
  {
    const int tpb = 256;
    f32_to_bf16_k<<<(int)((NQ + tpb - 1) / tpb), tpb, 0, stream>>>(query, qb, (int)NQ);
    f32_to_bf16_k<<<(int)((NQ + tpb - 1) / tpb), tpb, 0, stream>>>(keyv, kvb, (int)NQ);
    f32_to_bf16_k<<<(int)((NW + tpb - 1) / tpb), tpb, 0, stream>>>(Wq, wqb, (int)NW);
    f32_to_bf16_k<<<(int)((NW + tpb - 1) / tpb), tpb, 0, stream>>>(Wk, wkb, (int)NW);
    f32_to_bf16_k<<<(int)((NW + tpb - 1) / tpb), tpb, 0, stream>>>(Wv, wvb, (int)NW);
    f32_to_bf16_k<<<(int)((NW + tpb - 1) / tpb), tpb, 0, stream>>>(Wo, wob, (int)NW);
  }

  // 2) projections: X @ W^T + b  (bf16 out)
  {
    dim3 grid(D / 64, M / 256), block(128);
    gemm_bias_wmma<unsigned short><<<grid, block, 0, stream>>>(qb,  wqb, bq, Qp, M, D, D);
    gemm_bias_wmma<unsigned short><<<grid, block, 0, stream>>>(kvb, wkb, bk, Kp, M, D, D);
    gemm_bias_wmma<unsigned short><<<grid, block, 0, stream>>>(kvb, wvb, bv, Vp, M, D, D);
  }

  // 3) fused causal attention per (b,h,q-tile)
  {
    dim3 grid(S / 64, Bn * H), block(128);
    flash_attn_wmma<<<grid, block, 0, stream>>>(Qp, Kp, Vp, Oat, S, H, D);
  }

  // 4) output projection: Oat @ Wo^T + bo -> fp32 d_out
  {
    dim3 grid(D / 64, M / 256), block(128);
    gemm_bias_wmma<float><<<grid, block, 0, stream>>>(Oat, wob, bo, out, M, D, D);
  }
}
